// LSTM_27573690040569
// MI455X (gfx1250) — compile-verified, matrix operand order checked by reading, not run on hardware
//
#include <hip/hip_runtime.h>
#include <math.h>

typedef __attribute__((ext_vector_type(16))) _Float16 v16h;
typedef __attribute__((ext_vector_type(8)))  _Float16 v8h;
typedef __attribute__((ext_vector_type(8)))  float    v8f;

#define H    106
#define HP   112      // j padded to 7*16
#define NG   448      // 4*HP reordered gate columns [i|f|g|o]
#define K0   112      // layer0 K: x(6) + h0(106)
#define K1   224      // layer1 K: h0(112) + h1(112), exactly 7*32
#define BT   32       // batch tile per workgroup
#define SEQ  96
#define DIN  6
#define NTHREADS 448  // 14 waves: 2 m-tiles x 7 j-tiles

// LDS layout (offsets in _Float16 units unless noted)
#define OFF_B0 0
#define OFF_B1 (NG * K0)
#define OFF_X0 (OFF_B1 + NG * K1)
#define OFF_X1 (OFF_X0 + BT * K0)
#define OFF_END (OFF_X1 + BT * K1)
#define OFF_BIAS_BYTES (OFF_END * 2)
#define SMEM_BYTES (OFF_BIAS_BYTES + 2 * NG * 4)   // 326,144 bytes <= 320KB

__device__ __forceinline__ float sigm(float v) {
    return 1.0f / (1.0f + __expf(-v));
}

// Load a 16-bit A/B WMMA fragment chunk pair from LDS.
// p points at (row base + k0 + half*8); lo covers 8 halves, hi covers 8 halves at +16.
__device__ __forceinline__ v16h frag_ld(const _Float16* p) {
    v8h lo = *(const v8h*)p;
    v8h hi = *(const v8h*)(p + 16);
    v16h f;
#pragma unroll
    for (int i = 0; i < 8; ++i) { f[i] = lo[i]; f[8 + i] = hi[i]; }
    return f;
}

// Same, but upper K-chunk is zero (K tail padding done in registers, not LDS).
__device__ __forceinline__ v16h frag_ld_lo(const _Float16* p) {
    v8h lo = *(const v8h*)p;
    v16h f;
#pragma unroll
    for (int i = 0; i < 8; ++i) { f[i] = lo[i]; f[8 + i] = (_Float16)0.0f; }
    return f;
}

__global__ __launch_bounds__(NTHREADS, 1)
void lstm2_fused_wmma(const float* __restrict__ x,
                      const float* __restrict__ Wih0, const float* __restrict__ Whh0,
                      const float* __restrict__ bih0, const float* __restrict__ bhh0,
                      const float* __restrict__ Wih1, const float* __restrict__ Whh1,
                      const float* __restrict__ bih1, const float* __restrict__ bhh1,
                      const float* __restrict__ fcw,  const float* __restrict__ fcb,
                      float* __restrict__ out)
{
    extern __shared__ _Float16 smem[];
    _Float16* B0 = smem + OFF_B0;   // [NG][K0]  weights layer0 (f16)
    _Float16* B1 = smem + OFF_B1;   // [NG][K1]  weights layer1 (f16)
    _Float16* X0 = smem + OFF_X0;   // [BT][K0]  state: [x_t | h0]
    _Float16* X1 = smem + OFF_X1;   // [BT][K1]  state: [h0_t | h1]
    float* biasC = (float*)((char*)smem + OFF_BIAS_BYTES); // [2][NG]

    const int tid = threadIdx.x;
    const int b0  = blockIdx.x * BT;

    // ---------------- preamble: reorder + convert weights into LDS ----------------
    // reordered column n: gate gi = n/HP (i,f,g,o), j = n%HP; source row r = gi*H + j
    for (int e = tid; e < NG * K0; e += NTHREADS) {
        int n = e / K0, k = e % K0;
        int gi = n / HP, j = n % HP;
        float v = 0.0f;
        if (j < H) {
            int r = gi * H + j;
            if (k < DIN) v = Wih0[r * DIN + k];
            else         v = Whh0[r * H + (k - DIN)];
        }
        B0[e] = (_Float16)v;
    }
    for (int e = tid; e < NG * K1; e += NTHREADS) {
        int n = e / K1, k = e % K1;
        int gi = n / HP, j = n % HP;
        float v = 0.0f;
        if (j < H) {
            int r = gi * H + j;
            if (k < HP) { if (k < H) v = Wih1[r * H + k]; }
            else        { int km = k - HP; if (km < H) v = Whh1[r * H + km]; }
        }
        B1[e] = (_Float16)v;
    }
    for (int n = tid; n < NG; n += NTHREADS) {
        int gi = n / HP, j = n % HP;
        float v0 = 0.0f, v1 = 0.0f;
        if (j < H) { int r = gi * H + j; v0 = bih0[r] + bhh0[r]; v1 = bih1[r] + bhh1[r]; }
        biasC[n] = v0; biasC[NG + n] = v1;
    }
    for (int e = tid; e < BT * K0 + BT * K1; e += NTHREADS) {
        if (e < BT * K0) X0[e] = (_Float16)0.0f;
        else             X1[e - BT * K0] = (_Float16)0.0f;
    }
    __syncthreads();
    // stage x_0 into X0 columns [0..5]
    if (tid < BT * DIN) {
        int m = tid / DIN, d = tid % DIN;
        X0[m * K0 + d] = (_Float16)x[((size_t)(b0 + m) * SEQ + 0) * DIN + d];
    }
    __syncthreads();

    // ---------------- wave / lane mapping ----------------
    const int wave  = tid >> 5;
    const int lane  = tid & 31;
    const int half  = lane >> 4;     // K sub-chunk selector (WMMA 16-bit layout)
    const int l16   = lane & 15;     // M (A rows) / N (B,D cols)
    const int mtile = wave / 7;      // 0..1 : batch tile
    const int jtile = wave % 7;      // 0..6 : hidden-column tile (all 4 gates)
    const int jcol  = jtile * 16 + l16;

    float bs0[4], bs1[4];
#pragma unroll
    for (int gi = 0; gi < 4; ++gi) {
        bs0[gi] = biasC[gi * HP + jcol];
        bs1[gi] = biasC[NG + gi * HP + jcol];
    }

    const _Float16* x0row = X0 + (mtile * 16 + l16) * K0;  // A-fragment row base, layer 0
    const _Float16* x1row = X1 + (mtile * 16 + l16) * K1;  // A-fragment row base, layer 1

    v8f c0 = {};   // cell state layer0, this wave's [16 batch x 16 j] tile
    v8f c1 = {};   // cell state layer1

    // ---------------- recurrence over time ----------------
    for (int t = 0; t < SEQ; ++t) {
        // Layer-0 A fragments: [x_t | h0_{t-1}], K=112 (last slice zero-padded to 32)
        v16h a0[4];
#pragma unroll
        for (int ks = 0; ks < 3; ++ks) a0[ks] = frag_ld(x0row + ks * 32 + half * 8);
        a0[3] = frag_ld_lo(x0row + 96 + half * 8);
        __syncthreads();                              // (A) all X0 reads done

        // prefetch x_{t+1} (safe: X0[.,0..5] readers fenced by barrier A)
        if (t + 1 < SEQ && tid < BT * DIN) {
            int m = tid / DIN, d = tid % DIN;
            X0[m * K0 + d] = (_Float16)x[((size_t)(b0 + m) * SEQ + (t + 1)) * DIN + d];
        }

        // GEMM0: gates0[m, gi*HP + j] over K=112
        v8f acc[4];
#pragma unroll
        for (int gi = 0; gi < 4; ++gi) {
            v8f a = {};
            const _Float16* bb = B0 + (gi * HP + jtile * 16 + l16) * K0 + half * 8;
#pragma unroll
            for (int ks = 0; ks < 3; ++ks) {
                v16h b = frag_ld(bb + ks * 32);
                a = __builtin_amdgcn_wmma_f32_16x16x32_f16(false, a0[ks], false, b,
                                                           (short)0, a, false, false);
            }
            {
                v16h b = frag_ld_lo(bb + 96);
                a = __builtin_amdgcn_wmma_f32_16x16x32_f16(false, a0[3], false, b,
                                                           (short)0, a, false, false);
            }
            acc[gi] = a;
        }

        // layer-0 cell update (all four gates are wave-local by construction)
#pragma unroll
        for (int r = 0; r < 8; ++r) {
            float iv = sigm (acc[0][r] + bs0[0]);
            float fv = sigm (acc[1][r] + bs0[1]);
            float gv = tanhf(acc[2][r] + bs0[2]);
            float ov = sigm (acc[3][r] + bs0[3]);
            float c  = fv * c0[r] + iv * gv;
            c0[r]    = c;
            float h  = ov * tanhf(c);
            if (jcol < H) {                      // keep padding columns exactly zero
                int mrow = mtile * 16 + half * 8 + r;
                _Float16 hh = (_Float16)h;
                X0[mrow * K0 + DIN + jcol] = hh; // feeds layer0 recurrence at t+1
                X1[mrow * K1 + jcol]       = hh; // feeds layer1 input at t
            }
        }
        __syncthreads();                              // (B) h0 writes visible

        // Layer-1 A fragments: [h0_t | h1_{t-1}], K=224 = 7 slices
        v16h a1[7];
#pragma unroll
        for (int ks = 0; ks < 7; ++ks) a1[ks] = frag_ld(x1row + ks * 32 + half * 8);
        __syncthreads();                              // (C) all X1 reads done

        // GEMM1
#pragma unroll
        for (int gi = 0; gi < 4; ++gi) {
            v8f a = {};
            const _Float16* bb = B1 + (gi * HP + jtile * 16 + l16) * K1 + half * 8;
#pragma unroll
            for (int ks = 0; ks < 7; ++ks) {
                v16h b = frag_ld(bb + ks * 32);
                a = __builtin_amdgcn_wmma_f32_16x16x32_f16(false, a1[ks], false, b,
                                                           (short)0, a, false, false);
            }
            acc[gi] = a;
        }

        // layer-1 cell update
#pragma unroll
        for (int r = 0; r < 8; ++r) {
            float iv = sigm (acc[0][r] + bs1[0]);
            float fv = sigm (acc[1][r] + bs1[1]);
            float gv = tanhf(acc[2][r] + bs1[2]);
            float ov = sigm (acc[3][r] + bs1[3]);
            float c  = fv * c1[r] + iv * gv;
            c1[r]    = c;
            float h  = ov * tanhf(c);
            if (jcol < H) {
                int mrow = mtile * 16 + half * 8 + r;
                X1[mrow * K1 + HP + jcol] = (_Float16)h;  // h1 feeds its own recurrence
            }
        }
        __syncthreads();                              // (D) h1 + x prefetch visible
    }

    // ---------------- output: out[b] = relu(h1_last) . fc_w + fc_b ----------------
    if (tid < BT) {
        int m = tid;
        float a = fcb[0];
        for (int j = 0; j < H; ++j) {
            float hv = (float)X1[m * K1 + HP + j];
            hv = hv > 0.0f ? hv : 0.0f;
            a += hv * fcw[j];
        }
        out[b0 + m] = a;
    }
}

extern "C" void kernel_launch(void* const* d_in, const int* in_sizes, int n_in,
                              void* d_out, int out_size, void* d_ws, size_t ws_size,
                              hipStream_t stream) {
    (void)in_sizes; (void)n_in; (void)d_ws; (void)ws_size; (void)out_size;
    const float* x    = (const float*)d_in[0];
    const float* Wih0 = (const float*)d_in[1];
    const float* Whh0 = (const float*)d_in[2];
    const float* bih0 = (const float*)d_in[3];
    const float* bhh0 = (const float*)d_in[4];
    const float* Wih1 = (const float*)d_in[5];
    const float* Whh1 = (const float*)d_in[6];
    const float* bih1 = (const float*)d_in[7];
    const float* bhh1 = (const float*)d_in[8];
    const float* fcw  = (const float*)d_in[9];
    const float* fcb  = (const float*)d_in[10];
    float* out = (float*)d_out;

    dim3 grid(8192 / BT);       // 256 workgroups, one per WGP-resident LDS image
    dim3 block(NTHREADS);       // 14 waves (wave32)
    lstm2_fused_wmma<<<grid, block, SMEM_BYTES, stream>>>(
        x, Wih0, Whh0, bih0, bhh0, Wih1, Whh1, bih1, bhh1, fcw, fcb, out);
}
